// TransformerBlock_3D_1228360646684
// MI455X (gfx1250) — matrix operand order repaired and checked by reading, not compiled, not run on hardware
//
#include <hip/hip_runtime.h>
#include <math.h>
#include <stdint.h>

// ---------------------------------------------------------------------------
// CDNA5 (gfx1250) 3D deformable-attention transformer block.
// Channel-mixing convs = implicit GEMM on v_wmma_f32_16x16x32_f16 (wave32).
//  - activations padded to 64 channels -> unconditional K loads
//  - weights pre-packed into ISA 7.12.2 f16 A-fragment layout (32B/lane,
//    loaded as 2x global_load_b128)
//  - one wave owns a 16-voxel column block x full Cout (M-blocked, B reuse)
//  - deformable conv fully fused: trilinear sampling feeds the B fragment
//    directly; 27 taps accumulate in registers, one store of the result
//    (saves ~1.4 GB of DRAM round-trips vs. a sample/GEMM split).
// VALU kernels only for LayerNorm / depthwise convs (gather/bandwidth bound).
// ---------------------------------------------------------------------------

typedef __attribute__((ext_vector_type(16))) _Float16 v16h;
typedef __attribute__((ext_vector_type(8)))  float    v8f;

#define CCH   48
#define CPAD  64
#define SDIM  32
#define SVOX  (SDIM * SDIM * SDIM)   // 32768 = 1<<15
#define BVAL  2
#define NTOT  (BVAL * SVOX)          // 65536 voxel columns
#define NKS   2                      // K steps: 48 padded to 64 = 2 x 32

// epilogue modes
#define GM_BIAS   0   // out = gemm + bias
#define GM_GELU   1   // out = gelu(gemm + bias)
#define GM_P2FUSE 3   // out = x + gamma[m]*(gemm + bias + xn)
#define GM_ADDAUX 4   // out = skip + gemm + bias
#define GM_BNLRE  5   // out = leaky(bn(gemm))
#define GM_BNSKIP 6   // out = leaky(bn(gemm) + skip)

__device__ __forceinline__ float leakyf(float v) { return v >= 0.f ? v : 0.01f * v; }
__device__ __forceinline__ int   clampi(int v, int lo, int hi) { return v < lo ? lo : (v > hi ? hi : v); }

// A fragment K index for 16-bit A 16x32 (ISA 7.12.2), element 2j+e, lane half.
__device__ __forceinline__ int a_kidx(int j, int e, int half) {
    return (j < 4) ? (2 * j + e + 8 * half) : (16 + 2 * (j - 4) + e + 8 * half);
}

// ---------------------------------------------------------------------------
// Pack weights into A-fragment layout: frag = (mt*NKS + kki)*ntap + t,
// 32 lanes x 16 f16 (32B) per fragment. Zero-fills K>=Cin and m>=Cout.
// ---------------------------------------------------------------------------
__global__ void pack_weights(const float* __restrict__ W, v16h* __restrict__ packA,
                             int Cin, int Cout, int wstride, int ntap)
{
    const int wave = blockIdx.x * (blockDim.x >> 5) + (threadIdx.x >> 5);
    const int mtiles = (Cout + 15) >> 4;
    if (wave >= mtiles * NKS * ntap) return;
    const int t   = wave % ntap;
    const int kki = (wave / ntap) % NKS;
    const int mt  = wave / (ntap * NKS);
    const int l = threadIdx.x & 31;
    const int m = (mt << 4) + (l & 15);
    const int half = l >> 4;
    v16h a;
#pragma unroll
    for (int idx = 0; idx < 16; ++idx) {
        const int j = idx >> 1, e = idx & 1;
        const int K = (kki << 5) + a_kidx(j, e, half);
        a[idx] = (_Float16)((m < Cout && K < Cin)
                     ? W[((size_t)m * Cin + K) * wstride + t] : 0.f);
    }
    packA[(size_t)wave * 32 + l] = a;
}

// ---------------------------------------------------------------------------
// Unified implicit-GEMM kernel. One wave owns one 16-voxel column block and
// the full Cout extent (MT accumulator tiles).
//  NTAP=1             -> 1x1x1 conv
//  NTAP=27, SAMP=0    -> 3x3x3 conv pad 1 (halo via per-lane cndmask on B)
//  NTAP=27, SAMP=1    -> deformable 3x3x3: B = trilinear(in, offsets[tap])
// B input always CPAD(64)-channel padded; exactly NTOT/16 waves launched.
// ---------------------------------------------------------------------------
template<int COUT, int NTAP, int MODE, int OSTR, int SAMP = 0>
__global__ __launch_bounds__(256) void gemm_wmma(
    const float* __restrict__ in,        // [B, CPAD, SVOX]
    const v16h*  __restrict__ packA,
    int tapStride, int tapBase,
    const float* __restrict__ bias,
    float* __restrict__ out,             // [B, OSTR, SVOX]
    const float* __restrict__ aux0,      // P2FUSE: x(48) | ADDAUX/BNSKIP: skip(64)
    const float* __restrict__ aux1,      // P2FUSE: xn(64) | SAMP: offsets [B,81,S]
    const float* __restrict__ gv0,       // gamma | bn_s
    const float* __restrict__ gv1,       // bn_b
    const float* __restrict__ gv2,       // bn_m
    const float* __restrict__ gv3)       // bn_v
{
    constexpr int MT = (COUT + 15) >> 4;
    const int l = threadIdx.x & 31;
    const int wave = blockIdx.x * (blockDim.x >> 5) + (threadIdx.x >> 5);
    const int half = l >> 4;
    const int ncol = (wave << 4) | (l & 15);
    const int bidx = ncol >> 15;
    const int svox = ncol & (SVOX - 1);
    const float* inb = in + (size_t)bidx * CPAD * SVOX;
    float* ob = out + (size_t)bidx * OSTR * SVOX + svox;

    v8f acc[MT] = {};
    const int gx = svox & 31, gy = (svox >> 5) & 31, gz = svox >> 10;

    for (int t = 0; t < NTAP; ++t) {
        int s2 = svox; bool valid = true;
        int idxs[8]; float wts[8];
        if constexpr (SAMP) {
            // deformable tap: per-column fractional sample position
            const int kd = t / 9 - 1, kh = (t / 3) % 3 - 1, kw = t % 3 - 1;
            const float* offb = aux1 + ((size_t)bidx * 81 + t * 3) * SVOX + svox;
            __builtin_prefetch(offb + 3 * (size_t)SVOX, 0, 1);   // next tap's offsets
            const float z = (float)(gz + kd) + offb[0];
            const float y = (float)(gy + kh) + offb[(size_t)SVOX];
            const float x = (float)(gx + kw) + offb[(size_t)2 * SVOX];
            const float z0 = floorf(z), y0 = floorf(y), x0 = floorf(x);
            const float tz = z - z0, ty = y - y0, tx = x - x0;
            const int iz0 = (int)z0, iy0 = (int)y0, ix0 = (int)x0;
            int q = 0;
#pragma unroll
            for (int dz = 0; dz < 2; ++dz)
#pragma unroll
                for (int dy = 0; dy < 2; ++dy)
#pragma unroll
                    for (int dx = 0; dx < 2; ++dx) {
                        const int zi = iz0 + dz, yi = iy0 + dy, xi = ix0 + dx;
                        const float wgt = (dz ? tz : 1.f - tz) * (dy ? ty : 1.f - ty)
                                        * (dx ? tx : 1.f - tx);
                        const bool ok = (zi >= 0 && zi < 32 && yi >= 0 && yi < 32
                                         && xi >= 0 && xi < 32);
                        idxs[q] = (clampi(zi, 0, 31) << 10) + (clampi(yi, 0, 31) << 5)
                                  + clampi(xi, 0, 31);
                        wts[q] = ok ? wgt : 0.f;
                        ++q;
                    }
        } else if constexpr (NTAP > 1) {
            const int dz = t / 9 - 1, dy = (t / 3) % 3 - 1, dx = t % 3 - 1;
            const int zz = gz + dz, yy = gy + dy, xx = gx + dx;
            valid = ((unsigned)zz < 32u) & ((unsigned)yy < 32u) & ((unsigned)xx < 32u);
            s2 = valid ? ((zz << 10) + (yy << 5) + xx) : 0;
        }
        const float* inp = inb + s2;
#pragma unroll
        for (int kki = 0; kki < NKS; ++kki) {
            v16h bf;
#pragma unroll
            for (int idx = 0; idx < 16; ++idx) {
                const int K = (kki << 5) + ((idx < 8) ? (idx + (half << 3)) : (16 + (idx - 8) + (half << 3)));    // < CPAD always
                float v;
                if constexpr (SAMP) {
                    const float* ck = inb + (size_t)K * SVOX;    // pad channels are zero
                    v = 0.f;
#pragma unroll
                    for (int q2 = 0; q2 < 8; ++q2) v += wts[q2] * ck[idxs[q2]];
                } else {
                    v = inp[(size_t)K * SVOX];                   // unconditional load
                    if constexpr (NTAP > 1) v = valid ? v : 0.f; // halo zero (cndmask)
                }
                bf[idx] = (_Float16)v;
            }
#pragma unroll
            for (int mt = 0; mt < MT; ++mt) {
                const v16h a = packA[((size_t)((mt * NKS + kki) * tapStride + tapBase + t) << 5) + l];
                acc[mt] = __builtin_amdgcn_wmma_f32_16x16x32_f16(
                    false, a, false, bf, (short)0, acc[mt], false, false);
            }
        }
    }

#pragma unroll
    for (int mt = 0; mt < MT; ++mt)
#pragma unroll
        for (int i = 0; i < 8; ++i) {
            const int m = (mt << 4) + i + (half << 3);
            if (m >= COUT) continue;                 // compile-time except last 81-tile
            float v = acc[mt][i];
            if constexpr (MODE == GM_BIAS) {
                v += bias[m];
            } else if constexpr (MODE == GM_GELU) {
                v += bias[m];
                v = 0.5f * v * (1.f + erff(v * 0.70710678118654752f));
            } else if constexpr (MODE == GM_P2FUSE) {
                v += bias[m] + aux1[((size_t)bidx * CPAD + m) * SVOX + svox];
                v = aux0[((size_t)bidx * CCH + m) * SVOX + svox] + gv0[m] * v;
            } else if constexpr (MODE == GM_ADDAUX) {
                v += bias[m] + aux0[((size_t)bidx * CPAD + m) * SVOX + svox];
            } else if constexpr (MODE == GM_BNLRE || MODE == GM_BNSKIP) {
                const float inv = rsqrtf(gv3[m] + 1e-5f);
                v = (v - gv2[m]) * (gv0[m] * inv) + gv1[m];
                if constexpr (MODE == GM_BNSKIP)
                    v += aux0[((size_t)bidx * CPAD + m) * SVOX + svox];
                v = leakyf(v);
            }
            ob[(size_t)m * SVOX] = v;
        }
}

// --------------------------- VALU helper kernels ---------------------------

__global__ void zero_ws(float* __restrict__ w, size_t n)
{
    const size_t i = (size_t)blockIdx.x * blockDim.x + threadIdx.x;
    if (i < n) w[i] = 0.f;
}

__global__ void ln_kernel(const float* __restrict__ x, const float* __restrict__ s_,
                          const float* __restrict__ b_, float* __restrict__ xn)
{
    const int i = blockIdx.x * blockDim.x + threadIdx.x;
    if (i >= NTOT) return;
    const int b = i >> 15, s = i & (SVOX - 1);
    const float* xp = x + (size_t)b * CCH * SVOX + s;
    float mu = 0.f;
    for (int c = 0; c < CCH; ++c) mu += xp[(size_t)c * SVOX];
    mu *= (1.f / CCH);
    float var = 0.f;
    for (int c = 0; c < CCH; ++c) { const float d = xp[(size_t)c * SVOX] - mu; var += d * d; }
    var *= (1.f / CCH);
    const float inv = rsqrtf(var + 1e-5f);
    float* op = xn + (size_t)b * CPAD * SVOX + s;
    for (int c = 0; c < CCH; ++c)
        op[(size_t)c * SVOX] = (xp[(size_t)c * SVOX] - mu) * inv * s_[c] + b_[c];
}

__global__ void depthwise_kernel(const float* __restrict__ in, const float* __restrict__ w,
                                 const float* __restrict__ bias, float* __restrict__ out,
                                 int K, int pad, int dil)
{
    const int i = blockIdx.x * blockDim.x + threadIdx.x;
    if (i >= BVAL * CCH * SVOX) return;
    const int s = i & (SVOX - 1);
    const int c = (i >> 15) % CCH;
    const int b = i / (CCH * SVOX);
    const int x = s & 31, y = (s >> 5) & 31, z = s >> 10;
    const size_t base = ((size_t)b * CPAD + c) * SVOX;
    const float* inp = in + base;
    const float* wc = w + (size_t)c * K * K * K;
    float acc = bias[c];
    for (int kd = 0; kd < K; ++kd) {
        const int zz = z + kd * dil - pad;
        if ((unsigned)zz >= 32u) continue;
        for (int kh = 0; kh < K; ++kh) {
            const int yy = y + kh * dil - pad;
            if ((unsigned)yy >= 32u) continue;
            for (int kw = 0; kw < K; ++kw) {
                const int xx = x + kw * dil - pad;
                if ((unsigned)xx >= 32u) continue;
                acc += wc[(kd * K + kh) * K + kw] * inp[(zz << 10) + (yy << 5) + xx];
            }
        }
    }
    out[base + s] = acc;
}

__global__ void mul_kernel(const float* __restrict__ a, const float* __restrict__ b,
                           float* __restrict__ o)
{
    const int i = blockIdx.x * blockDim.x + threadIdx.x;
    if (i >= BVAL * CCH * SVOX) return;
    const int s = i & (SVOX - 1);
    const int c = (i >> 15) % CCH;
    const int bb = i / (CCH * SVOX);
    const size_t off = ((size_t)bb * CPAD + c) * SVOX + s;
    o[off] = a[off] * b[off];
}

// ---------------------------------------------------------------------------
extern "C" void kernel_launch(void* const* d_in, const int* in_sizes, int n_in,
                              void* d_out, int out_size, void* d_ws, size_t ws_size,
                              hipStream_t stream)
{
    (void)in_sizes; (void)n_in; (void)out_size; (void)ws_size;
    const float* x     = (const float*)d_in[0];
    const float* ln_s  = (const float*)d_in[1];
    const float* ln_b  = (const float*)d_in[2];
    const float* gamma = (const float*)d_in[3];
    const float* p1_w  = (const float*)d_in[4];
    const float* p1_b  = (const float*)d_in[5];
    const float* p2_w  = (const float*)d_in[6];
    const float* p2_b  = (const float*)d_in[7];
    const float* c0_w  = (const float*)d_in[8];
    const float* c0_b  = (const float*)d_in[9];
    const float* cs_w  = (const float*)d_in[10];
    const float* cs_b  = (const float*)d_in[11];
    const float* off_w = (const float*)d_in[12];
    const float* off_b = (const float*)d_in[13];
    const float* dc_w  = (const float*)d_in[14];
    const float* dc_b  = (const float*)d_in[15];
    const float* c1_w  = (const float*)d_in[16];
    const float* c1_b  = (const float*)d_in[17];
    const float* u1_w  = (const float*)d_in[18];
    const float* bn1_s = (const float*)d_in[19];
    const float* bn1_b = (const float*)d_in[20];
    const float* bn1_m = (const float*)d_in[21];
    const float* bn1_v = (const float*)d_in[22];
    const float* u2_w  = (const float*)d_in[23];
    const float* bn2_s = (const float*)d_in[24];
    const float* bn2_b = (const float*)d_in[25];
    const float* bn2_m = (const float*)d_in[26];
    const float* bn2_v = (const float*)d_in[27];
    const float* pr_w  = (const float*)d_in[28];
    const float* pr_b  = (const float*)d_in[29];
    float* outp = (float*)d_out;

    float* wsf = (float*)d_ws;
    const size_t E64 = (size_t)BVAL * CPAD * SVOX;   // 4,194,304 floats / buffer
    float* xn    = wsf + 0 * E64;
    float* v1    = wsf + 1 * E64;    // p1+gelu output (gating "u")
    float* t0    = wsf + 2 * E64;    // dw5 out; reused as v2 = u*a
    float* t1    = wsf + 3 * E64;    // dw7 out (deform input); reused a2, then h
    float* offt  = wsf + 4 * E64;    // [B,81,S], fits in slots 4..5
    float* dco   = wsf + 6 * E64;    // deform conv output; reused as attn
    float* skipb = wsf + 7 * E64;
    float* a2 = t1; float* v2 = t0; float* hb = t1; float* attn = dco;

    // packed-weight region (32B-aligned fragments, 1KB each)
    uint8_t* pb = (uint8_t*)(wsf + 8 * E64);
    v16h* pk_p1 = (v16h*)(pb + 0);          // 3*2*1  frags =   6KB
    v16h* pk_p2 = (v16h*)(pb + 8  * 1024);
    v16h* pk_c1 = (v16h*)(pb + 16 * 1024);
    v16h* pk_pr = (v16h*)(pb + 24 * 1024);
    v16h* pk_u1 = (v16h*)(pb + 32 * 1024);  // 3*2*27 frags = 162KB
    v16h* pk_u2 = (v16h*)(pb + 208 * 1024);
    v16h* pk_dc = (v16h*)(pb + 384 * 1024);
    v16h* pk_of = (v16h*)(pb + 560 * 1024); // 6*2*27 frags = 324KB

    const dim3 blk(256);
    const int gN = (NTOT + 255) / 256;
    const int gE = (BVAL * CCH * SVOX + 255) / 256;
    const int gGemm = (NTOT / 16) / 8;                       // 4096 waves exactly
    auto pack_blocks = [](int Cout, int ntap) {
        return (((Cout + 15) / 16) * NKS * ntap + 7) / 8;
    };

    // 0. zero padded workspace (pad channels must be numerically clean)
    {
        const size_t n = 8 * E64;
        zero_ws<<<(int)((n + 255) / 256), blk, 0, stream>>>(wsf, n);
    }
    // 0b. pack all weights into f16 A-fragment layout
    pack_weights<<<pack_blocks(48, 1),  blk, 0, stream>>>(p1_w,  pk_p1, 48, 48, 1,  1);
    pack_weights<<<pack_blocks(48, 1),  blk, 0, stream>>>(p2_w,  pk_p2, 48, 48, 1,  1);
    pack_weights<<<pack_blocks(48, 1),  blk, 0, stream>>>(c1_w,  pk_c1, 48, 48, 1,  1);
    pack_weights<<<pack_blocks(48, 1),  blk, 0, stream>>>(pr_w,  pk_pr, 48, 48, 1,  1);
    pack_weights<<<pack_blocks(48, 27), blk, 0, stream>>>(u1_w,  pk_u1, 48, 48, 27, 27);
    pack_weights<<<pack_blocks(48, 27), blk, 0, stream>>>(u2_w,  pk_u2, 48, 48, 27, 27);
    pack_weights<<<pack_blocks(48, 27), blk, 0, stream>>>(dc_w,  pk_dc, 48, 48, 27, 27);
    pack_weights<<<pack_blocks(81, 27), blk, 0, stream>>>(off_w, pk_of, 48, 81, 27, 27);

    // 1. LayerNorm over channels (into padded layout)
    ln_kernel<<<gN, blk, 0, stream>>>(x, ln_s, ln_b, xn);
    // 2. proj_1 + exact GELU
    gemm_wmma<48, 1, GM_GELU, CPAD><<<gGemm, blk, 0, stream>>>(
        xn, pk_p1, 1, 0, p1_b, v1, nullptr, nullptr, nullptr, nullptr, nullptr, nullptr);
    // 3. depthwise 5x5x5 pad 2
    depthwise_kernel<<<gE, blk, 0, stream>>>(v1, c0_w, c0_b, t0, 5, 2, 1);
    // 4. depthwise 7x7x7 dil 3 pad 9
    depthwise_kernel<<<gE, blk, 0, stream>>>(t0, cs_w, cs_b, t1, 7, 9, 3);
    // 5. offset conv 3x3x3: 48 -> 81
    gemm_wmma<81, 27, GM_BIAS, 81><<<gGemm, blk, 0, stream>>>(
        t1, pk_of, 27, 0, off_b, offt, nullptr, nullptr, nullptr, nullptr, nullptr, nullptr);
    // 6. deformable conv, fully fused: trilinear B sampling + 27-tap register
    //    accumulation + bias, one store.
    gemm_wmma<48, 27, GM_BIAS, CPAD, 1><<<gGemm, blk, 0, stream>>>(
        t1, pk_dc, 27, 0, dc_b, dco, nullptr, offt, nullptr, nullptr, nullptr, nullptr);
    // 7. c1 (1x1x1)
    gemm_wmma<48, 1, GM_BIAS, CPAD><<<gGemm, blk, 0, stream>>>(
        dco, pk_c1, 1, 0, c1_b, a2, nullptr, nullptr, nullptr, nullptr, nullptr, nullptr);
    // 8. spatial gating: v2 = u * a
    mul_kernel<<<gE, blk, 0, stream>>>(v1, a2, v2);
    // 9. proj_2 fused: skip = x + gamma*(p2@v2 + b + xn)
    gemm_wmma<48, 1, GM_P2FUSE, CPAD><<<gGemm, blk, 0, stream>>>(
        v2, pk_p2, 1, 0, p2_b, skipb, x, xn, gamma, nullptr, nullptr, nullptr);
    // 10. UnetResBlock conv1: leaky(bn1(conv3(skip)))
    gemm_wmma<48, 27, GM_BNLRE, CPAD><<<gGemm, blk, 0, stream>>>(
        skipb, pk_u1, 27, 0, nullptr, hb, nullptr, nullptr, bn1_s, bn1_b, bn1_m, bn1_v);
    // 11. UnetResBlock conv2: attn = leaky(bn2(conv3(h)) + skip)
    gemm_wmma<48, 27, GM_BNSKIP, CPAD><<<gGemm, blk, 0, stream>>>(
        hb, pk_u2, 27, 0, nullptr, attn, skipb, nullptr, bn2_s, bn2_b, bn2_m, bn2_v);
    // 12. projconv fused: out = skip + pr@attn + pr_b  (unpadded 48-ch output)
    gemm_wmma<48, 1, GM_ADDAUX, CCH><<<gGemm, blk, 0, stream>>>(
        attn, pk_pr, 1, 0, pr_b, outp, skipb, nullptr, nullptr, nullptr, nullptr, nullptr);
}
